// TokenPoseWithShiftedWindow_26671746908308
// MI455X (gfx1250) — matrix operand order repaired and checked by reading, not compiled, not run on hardware
//
#include <hip/hip_runtime.h>
#include <stdint.h>

// ---------------------------------------------------------------------------
// MI455X / gfx1250: all heavy GEMMs on v_wmma_f32_16x16x32_bf16 (f32 accum).
// bf16 weight arena stored in WMMA B-fragment order -> B fragments load
// directly from global as 2x global_load_b128; A tiles staged in LDS so A
// fragments are 2x ds_load_b128. Epilogues specialized at compile time.
// ---------------------------------------------------------------------------

typedef __attribute__((ext_vector_type(16))) __bf16 v16bf;
typedef __attribute__((ext_vector_type(8)))  float  v8f;

#define LN_EPS 1e-5f

union Frag { v16bf v; uint4 q[2]; unsigned short u[16]; };

__device__ __forceinline__ unsigned short f2bf(float f) {
  union { float f; unsigned u; } c; c.f = f;
  unsigned u = c.u;
  u += 0x7fffu + ((u >> 16) & 1u);      // round-to-nearest-even
  return (unsigned short)(u >> 16);
}
__device__ __forceinline__ unsigned pk2(float a, float b) {
  return (unsigned)f2bf(a) | ((unsigned)f2bf(b) << 16);
}
__device__ __forceinline__ uint4 pk8(const float4 x0, const float4 x1) {
  uint4 p;
  p.x = pk2(x0.x, x0.y); p.y = pk2(x0.z, x0.w);
  p.z = pk2(x1.x, x1.y); p.w = pk2(x1.z, x1.w);
  return p;
}

// A fragment from LDS tile [row][k], stride 40 u16 (80B, 16B aligned).
__device__ __forceinline__ v16bf frag_a_lds(const unsigned short* rowp, int hi) {
  Frag f;
  f.q[0] = *(const uint4*)(rowp + (hi << 4));
  f.q[1] = *(const uint4*)(rowp + (hi << 4) + 8);
  return f.v;
}
// B fragment from transposed LDS tile [col][k] (stride 40)
__device__ __forceinline__ v16bf frag_b_ldsT(const unsigned short* colp, int hi) {
  Frag f;
  f.q[0] = *(const uint4*)(colp + (hi << 4));
  f.q[1] = *(const uint4*)(colp + (hi << 4) + 8);
  return f.v;
}
// B fragment direct from weight arena block (512 u16 = [col(16)][k(32)])
__device__ __forceinline__ v16bf frag_b_gbl(const unsigned short* blk, int lo, int hi) {
  const uint4* p = (const uint4*)(blk + (lo << 5) + (hi << 4));
  Frag f; f.q[0] = p[0]; f.q[1] = p[1];
  return f.v;
}

// Weight arena index: [ntile][kchunk] blocks of 512, block = [col][k]
__device__ __forceinline__ size_t warena_idx(int k, int n, int K) {
  return ((size_t)((n >> 4) * (K >> 5) + (k >> 5)) << 9) + ((size_t)(n & 15) << 5) + (k & 31);
}

// ---------------------------------------------------------------------------
// Weight conversion kernels (f32 -> bf16 arena, fragment-ordered)
// ---------------------------------------------------------------------------
__global__ __launch_bounds__(256) void cvt_mat_k(const float* __restrict__ src,
                                                 unsigned short* __restrict__ dst,
                                                 int K, int N) {
  int i = blockIdx.x * 256 + threadIdx.x;
  if (i >= K * N) return;
  int k = i / N, n = i - k * N;
  dst[warena_idx(k, n, K)] = f2bf(src[i]);
}

// patch_w [384, 512] -> arena K=512, N=384
__global__ __launch_bounds__(256) void cvt_patchT_k(const float* __restrict__ pw,
                                                    unsigned short* __restrict__ dst) {
  int i = blockIdx.x * 256 + threadIdx.x;
  if (i >= 512 * 384) return;
  int k = i / 384, n = i - k * 384;
  dst[warena_idx(k, n, 512)] = f2bf(pw[(size_t)n * 512 + k]);
}

// dense per-head Q/K/V [H,C,HD] x3 -> arena K=384, N=1152 col = s*384+h*64+d
__global__ __launch_bounds__(256) void cvt_qkv3_k(const float* __restrict__ qw,
                                                  const float* __restrict__ kw,
                                                  const float* __restrict__ vw,
                                                  unsigned short* __restrict__ dst) {
  int i = blockIdx.x * 256 + threadIdx.x;
  if (i >= 384 * 1152) return;
  int c = i / 1152, col = i - c * 1152;
  int s = col / 384, r = col - s * 384;
  int h = r >> 6, d = r & 63;
  const float* src = (s == 0) ? qw : ((s == 1) ? kw : vw);
  dst[warena_idx(c, col, 384)] = f2bf(src[((size_t)h * 384 + c) * 64 + d]);
}

// deconv w [C,O,2,2,2] -> arena K=384, N=3072, col = ijk*384 + o
__global__ __launch_bounds__(256) void cvt_deconv_k(const float* __restrict__ w,
                                                    unsigned short* __restrict__ dst) {
  int i = blockIdx.x * 256 + threadIdx.x;
  if (i >= 384 * 3072) return;
  int c = i / 3072, n = i - c * 3072;
  int ijk = n / 384, o = n - ijk * 384;
  dst[warena_idx(c, n, 384)] = f2bf(w[((size_t)c * 384 + o) * 8 + ijk]);
}

// ---------------------------------------------------------------------------
// im2col for patch embedding: x[B,1,64,64,64] -> [2048, 512]
// ---------------------------------------------------------------------------
__global__ __launch_bounds__(256) void im2col_k(const float* __restrict__ x,
                                                float* __restrict__ im) {
  int row = blockIdx.x;                    // b*512 + n
  int b = row >> 9, n = row & 511;
  int i0 = n >> 6, i1 = (n >> 3) & 7, i2 = n & 7;
  const float* xb = x + (size_t)b * 262144;
  for (int k = threadIdx.x; k < 512; k += 256) {
    int kz = k >> 6, ky = (k >> 3) & 7, kx = k & 7;
    im[(size_t)row * 512 + k] =
        xb[(size_t)(i0 * 8 + kz) * 4096 + (i1 * 8 + ky) * 64 + (i2 * 8 + kx)];
  }
}

// ---------------------------------------------------------------------------
// Generic WMMA GEMM, tile 64x128, 8 waves (each wave: 1 A frag -> 4 WMMAs).
// Compile-time epilogue: RESMODE 0 none / 1 full / 2 pos-emb broadcast.
// ---------------------------------------------------------------------------
template <int RESMODE, int RELU, int HASBIAS>
__global__ __launch_bounds__(256) void gemm_t(
    const float* __restrict__ A, int lda,
    const unsigned short* __restrict__ W,
    const float* __restrict__ bias,
    const float* __restrict__ res, int ldr,
    float* __restrict__ out, int ldo, int K) {
  __shared__ unsigned short sA[64][40];

  int tid = threadIdx.x;
  int nBase = blockIdx.x * 128;
  int rowBase = blockIdx.y * 64;
  int wave = tid >> 5, lane = tid & 31;
  int lo = lane & 15, hi = lane >> 4;
  int mi = wave >> 1, njBase = (wave & 1) * 4;
  int KC = K >> 5;
  int ar = tid >> 2, ac = (tid & 3) * 8;

  v8f acc[4] = {};
  for (int kc = 0; kc < KC; ++kc) {
    const float4* ap = (const float4*)(A + (size_t)(rowBase + ar) * lda + (kc << 5) + ac);
    float4 x0 = ap[0], x1 = ap[1];
    *(uint4*)&sA[ar][ac] = pk8(x0, x1);
    __syncthreads();
    v16bf a = frag_a_lds(&sA[mi * 16 + lo][0], hi);
#pragma unroll
    for (int t = 0; t < 4; ++t) {
      int ntile = (nBase >> 4) + njBase + t;
      v16bf b = frag_b_gbl(W + (((size_t)ntile * KC + kc) << 9), lo, hi);
      acc[t] = __builtin_amdgcn_wmma_f32_16x16x32_bf16(false, a, false, b,
                                                       (short)0, acc[t], false, false);
    }
    __syncthreads();
  }

#pragma unroll
  for (int t = 0; t < 4; ++t) {
    int gcol = nBase + (njBase + t) * 16 + lo;
    float bs = HASBIAS ? bias[gcol] : 0.0f;
#pragma unroll
    for (int r = 0; r < 8; ++r) {
      int grow = rowBase + mi * 16 + hi * 8 + r;
      float v = acc[t][r] + bs;
      if (RESMODE == 1) v += res[(size_t)grow * ldr + gcol];
      if (RESMODE == 2) v += res[(size_t)(grow & 511) * ldr + gcol];
      if (RELU) v = fmaxf(v, 0.0f);
      out[(size_t)grow * ldo + gcol] = v;
    }
  }
}

// ---------------------------------------------------------------------------
// Attention scores: S[bh,n,m] = (Q.K^T)/8 ; per (b,h) 512x512, K=64
// ---------------------------------------------------------------------------
__global__ __launch_bounds__(256) void attn_qk_k(const float* __restrict__ qkv,
                                                 float* __restrict__ S) {
  __shared__ unsigned short sQ[64][40];
  __shared__ unsigned short sKt[64][40];
  int tid = threadIdx.x;
  int bz = blockIdx.z;
  int b = bz / 6, h = bz % 6;
  int nBase = blockIdx.x * 64;      // keys
  int rowBase = blockIdx.y * 64;    // queries
  int wave = tid >> 5, lane = tid & 31;
  int lo = lane & 15, hi = lane >> 4;
  int mi = wave >> 1, njBase = (wave & 1) * 2;
  int ar = tid >> 2, ac = (tid & 3) * 8;

  v8f acc[2] = {};
  for (int kc = 0; kc < 2; ++kc) {
    {
      const float4* qp = (const float4*)(qkv + (size_t)(b * 512 + rowBase + ar) * 1152 +
                                         h * 64 + (kc << 5) + ac);
      *(uint4*)&sQ[ar][ac] = pk8(qp[0], qp[1]);
      const float4* kp = (const float4*)(qkv + (size_t)(b * 512 + nBase + ar) * 1152 +
                                         384 + h * 64 + (kc << 5) + ac);
      *(uint4*)&sKt[ar][ac] = pk8(kp[0], kp[1]);
    }
    __syncthreads();
    v16bf a = frag_a_lds(&sQ[mi * 16 + lo][0], hi);
#pragma unroll
    for (int t = 0; t < 2; ++t) {
      v16bf bf = frag_b_ldsT(&sKt[(njBase + t) * 16 + lo][0], hi);
      acc[t] = __builtin_amdgcn_wmma_f32_16x16x32_bf16(false, a, false, bf,
                                                       (short)0, acc[t], false, false);
    }
    __syncthreads();
  }
#pragma unroll
  for (int t = 0; t < 2; ++t) {
    int gcol = nBase + (njBase + t) * 16 + lo;
#pragma unroll
    for (int r = 0; r < 8; ++r) {
      int grow = rowBase + mi * 16 + hi * 8 + r;
      S[((size_t)bz * 512 + grow) * 512 + gcol] = acc[t][r] * 0.125f;
    }
  }
}

// softmax over 512-wide rows; one wave per row
__global__ __launch_bounds__(256) void softmax512_k(float* __restrict__ S) {
  int row = blockIdx.x * 8 + (threadIdx.x >> 5);
  int lane = threadIdx.x & 31;
  float* p = S + (size_t)row * 512;
  float v[16], mx = -3.4e38f;
#pragma unroll
  for (int i = 0; i < 16; ++i) { v[i] = p[lane + i * 32]; mx = fmaxf(mx, v[i]); }
  for (int off = 16; off; off >>= 1) mx = fmaxf(mx, __shfl_xor(mx, off, 32));
  float sum = 0.f;
#pragma unroll
  for (int i = 0; i < 16; ++i) { v[i] = __expf(v[i] - mx); sum += v[i]; }
  for (int off = 16; off; off >>= 1) sum += __shfl_xor(sum, off, 32);
  float inv = 1.0f / sum;
#pragma unroll
  for (int i = 0; i < 16; ++i) p[lane + i * 32] = v[i] * inv;
}

// O = P * V : per (b,h), M=512, N=64, K=512 ; out col = h*64 + d
__global__ __launch_bounds__(256) void attn_pv_k(const float* __restrict__ S,
                                                 const float* __restrict__ qkv,
                                                 float* __restrict__ attO) {
  __shared__ unsigned short sP[64][40];
  __shared__ unsigned short sVt[64][40];   // [d][m]
  int tid = threadIdx.x;
  int bz = blockIdx.y;
  int b = bz / 6, h = bz % 6;
  int rowBase = blockIdx.x * 64;
  const float* P = S + (size_t)bz * 512 * 512;
  int wave = tid >> 5, lane = tid & 31;
  int lo = lane & 15, hi = lane >> 4;
  int mi = wave >> 1, njBase = (wave & 1) * 2;
  int ar = tid >> 2, ac = (tid & 3) * 8;
  int vm = tid >> 3, vd = (tid & 7) * 8;

  v8f acc[2] = {};
  for (int kc = 0; kc < 16; ++kc) {
    {
      const float4* ap = (const float4*)(P + (size_t)(rowBase + ar) * 512 + (kc << 5) + ac);
      *(uint4*)&sP[ar][ac] = pk8(ap[0], ap[1]);
      const float4* vp = (const float4*)(qkv + (size_t)(b * 512 + (kc << 5) + vm) * 1152 +
                                         768 + h * 64 + vd);
      float4 y0 = vp[0], y1 = vp[1];
      sVt[vd + 0][vm] = f2bf(y0.x); sVt[vd + 1][vm] = f2bf(y0.y);
      sVt[vd + 2][vm] = f2bf(y0.z); sVt[vd + 3][vm] = f2bf(y0.w);
      sVt[vd + 4][vm] = f2bf(y1.x); sVt[vd + 5][vm] = f2bf(y1.y);
      sVt[vd + 6][vm] = f2bf(y1.z); sVt[vd + 7][vm] = f2bf(y1.w);
    }
    __syncthreads();
    v16bf a = frag_a_lds(&sP[mi * 16 + lo][0], hi);
#pragma unroll
    for (int t = 0; t < 2; ++t) {
      v16bf bf = frag_b_ldsT(&sVt[(njBase + t) * 16 + lo][0], hi);
      acc[t] = __builtin_amdgcn_wmma_f32_16x16x32_bf16(false, a, false, bf,
                                                       (short)0, acc[t], false, false);
    }
    __syncthreads();
  }
#pragma unroll
  for (int t = 0; t < 2; ++t) {
    int gcol = (njBase + t) * 16 + lo;   // d in [0,64)
#pragma unroll
    for (int r = 0; r < 8; ++r) {
      int grow = rowBase + mi * 16 + hi * 8 + r;
      attO[((size_t)b * 512 + grow) * 384 + h * 64 + gcol] = acc[t][r];
    }
  }
}

// ---------------------------------------------------------------------------
// LayerNorm over C=384 ; one wave per row
// ---------------------------------------------------------------------------
__global__ __launch_bounds__(256) void layernorm_k(const float* __restrict__ in,
                                                   const float* __restrict__ g,
                                                   const float* __restrict__ b,
                                                   float* __restrict__ out) {
  int row = blockIdx.x * 8 + (threadIdx.x >> 5);
  int lane = threadIdx.x & 31;
  const float* p = in + (size_t)row * 384;
  float v[12], s = 0.f;
#pragma unroll
  for (int i = 0; i < 12; ++i) { v[i] = p[lane + i * 32]; s += v[i]; }
  for (int off = 16; off; off >>= 1) s += __shfl_xor(s, off, 32);
  float mean = s * (1.0f / 384.0f);
  float q = 0.f;
#pragma unroll
  for (int i = 0; i < 12; ++i) { float d = v[i] - mean; q += d * d; }
  for (int off = 16; off; off >>= 1) q += __shfl_xor(q, off, 32);
  float inv = rsqrtf(q * (1.0f / 384.0f) + LN_EPS);
  float* o = out + (size_t)row * 384;
#pragma unroll
  for (int i = 0; i < 12; ++i) {
    int c = lane + i * 32;
    o[c] = (v[i] - mean) * inv * g[c] + b[c];
  }
}

// ---------------------------------------------------------------------------
// Shifted-window attention (8-token windows): VALU/LDS, 1 wave = (win,head)
// ---------------------------------------------------------------------------
__global__ __launch_bounds__(256) void swin_attn_k(const float* __restrict__ qkv,
                                                   float* __restrict__ attO) {
  __shared__ float sQ[8][8][64];
  __shared__ float sK[8][8][64];
  __shared__ float sV[8][8][64];
  __shared__ float sS[8][8][8];
  __shared__ int   sMap[8][8];
  int wave = threadIdx.x >> 5, lane = threadIdx.x & 31;
  int unit = blockIdx.x * 8 + wave;        // 0..1535
  int h = unit % 6;
  int win = unit / 6;                      // 0..255
  int b = win >> 6;
  int wr = win & 63;
  int wh = wr >> 4, ww = (wr >> 2) & 3, wd = wr & 3;

  if (lane < 8) {
    int ih = lane >> 2, iw = (lane >> 1) & 1, id = lane & 1;
    int oh = (wh * 2 + ih + 1) & 7;        // roll by -SHIFT
    int ow = (ww * 2 + iw + 1) & 7;
    int od = (wd * 2 + id + 1) & 7;
    sMap[wave][lane] = b * 512 + oh * 64 + ow * 8 + od;
  }
  __syncthreads();

  for (int t = 0; t < 8; ++t) {
    const float* base = qkv + (size_t)sMap[wave][t] * 1152 + h * 64;
    for (int d = lane; d < 64; d += 32) {
      sQ[wave][t][d] = base[d];
      sK[wave][t][d] = base[384 + d];
      sV[wave][t][d] = base[768 + d];
    }
  }
  __syncthreads();

  for (int p = lane; p < 64; p += 32) {
    int i = p >> 3, j = p & 7;
    float s = 0.f;
#pragma unroll
    for (int d = 0; d < 64; ++d) s += sQ[wave][i][d] * sK[wave][j][d];
    sS[wave][i][j] = s * 0.125f;
  }
  __syncthreads();

  if (lane < 8) {
    float mx = -3.4e38f;
#pragma unroll
    for (int j = 0; j < 8; ++j) mx = fmaxf(mx, sS[wave][lane][j]);
    float e[8], sum = 0.f;
#pragma unroll
    for (int j = 0; j < 8; ++j) { e[j] = __expf(sS[wave][lane][j] - mx); sum += e[j]; }
    float inv = 1.0f / sum;
#pragma unroll
    for (int j = 0; j < 8; ++j) sS[wave][lane][j] = e[j] * inv;
  }
  __syncthreads();

  for (int idx = lane; idx < 512; idx += 32) {
    int i = idx >> 6, d = idx & 63;
    float s = 0.f;
#pragma unroll
    for (int j = 0; j < 8; ++j) s += sS[wave][i][j] * sV[wave][j][d];
    attO[(size_t)sMap[wave][i] * 384 + h * 64 + d] = s;  // roll-back == same map
  }
}

// ---------------------------------------------------------------------------
// deconv1 (k=s=2): WMMA GEMM + BN1 + ReLU, scatter to spatial-major r1
// ---------------------------------------------------------------------------
__global__ __launch_bounds__(256) void deconv1_k(
    const float* __restrict__ A, const unsigned short* __restrict__ W,
    const float* __restrict__ cb, const float* __restrict__ g,
    const float* __restrict__ bb, const float* __restrict__ mn,
    const float* __restrict__ vr, float* __restrict__ r1) {
  __shared__ unsigned short sA[16][40];
  int tid = threadIdx.x;
  int rowBase = blockIdx.x * 16;
  int ijk = blockIdx.y;
  int i2 = (ijk >> 2) & 1, j2 = (ijk >> 1) & 1, k2 = ijk & 1;
  int wave = tid >> 5, lane = tid & 31;
  int lo = lane & 15, hi = lane >> 4;

  v8f acc[3] = {};
  for (int kc = 0; kc < 12; ++kc) {
    if (tid < 64) {
      int r = tid >> 2, c0 = (tid & 3) * 8;
      const float4* ap = (const float4*)(A + (size_t)(rowBase + r) * 384 + (kc << 5) + c0);
      *(uint4*)&sA[r][c0] = pk8(ap[0], ap[1]);
    }
    __syncthreads();
    v16bf a = frag_a_lds(&sA[lo][0], hi);
#pragma unroll
    for (int t = 0; t < 3; ++t) {
      int ntile = ijk * 24 + wave * 3 + t;
      v16bf bf = frag_b_gbl(W + (((size_t)ntile * 12 + kc) << 9), lo, hi);
      acc[t] = __builtin_amdgcn_wmma_f32_16x16x32_bf16(false, a, false, bf,
                                                       (short)0, acc[t], false, false);
    }
    __syncthreads();
  }
#pragma unroll
  for (int t = 0; t < 3; ++t) {
    int o = (wave * 3 + t) * 16 + lo;
    float db = cb[o];
    float sc = g[o] * rsqrtf(vr[o] + LN_EPS);
    float sh = bb[o] - mn[o] * sc;
#pragma unroll
    for (int r = 0; r < 8; ++r) {
      int grow = rowBase + hi * 8 + r;
      float val = fmaxf((acc[t][r] + db) * sc + sh, 0.0f);
      int b = grow >> 9, n = grow & 511;
      int hh = n >> 6, w2 = (n >> 3) & 7, dd = n & 7;
      size_t ri = (size_t)b * 4096 + (2 * hh + i2) * 256 + (2 * w2 + j2) * 16 + (2 * dd + k2);
      r1[ri * 384 + o] = val;
    }
  }
}

// ---------------------------------------------------------------------------
// deconv2 + BN2 + ReLU + final 1x1 conv (out_w [15,384]) fused -> d_out
// ---------------------------------------------------------------------------
__global__ __launch_bounds__(256) void deconv2_out_k(
    const float* __restrict__ A, const unsigned short* __restrict__ W,
    const float* __restrict__ cb, const float* __restrict__ g,
    const float* __restrict__ bb, const float* __restrict__ mn,
    const float* __restrict__ vr, const float* __restrict__ ow,
    const float* __restrict__ ob, float* __restrict__ out) {
  __shared__ unsigned short sA[16][40];
  __shared__ float sT[16][385];
  int tid = threadIdx.x;
  int rowBase = blockIdx.x * 16;
  int ijk = blockIdx.y;
  int i2 = (ijk >> 2) & 1, j2 = (ijk >> 1) & 1, k2 = ijk & 1;
  int wave = tid >> 5, lane = tid & 31;
  int lo = lane & 15, hi = lane >> 4;

  v8f acc[3] = {};
  for (int kc = 0; kc < 12; ++kc) {
    if (tid < 64) {
      int r = tid >> 2, c0 = (tid & 3) * 8;
      const float4* ap = (const float4*)(A + (size_t)(rowBase + r) * 384 + (kc << 5) + c0);
      *(uint4*)&sA[r][c0] = pk8(ap[0], ap[1]);
    }
    __syncthreads();
    v16bf a = frag_a_lds(&sA[lo][0], hi);
#pragma unroll
    for (int t = 0; t < 3; ++t) {
      int ntile = ijk * 24 + wave * 3 + t;
      v16bf bf = frag_b_gbl(W + (((size_t)ntile * 12 + kc) << 9), lo, hi);
      acc[t] = __builtin_amdgcn_wmma_f32_16x16x32_bf16(false, a, false, bf,
                                                       (short)0, acc[t], false, false);
    }
    __syncthreads();
  }
  // BN2 + ReLU into LDS tile
#pragma unroll
  for (int t = 0; t < 3; ++t) {
    int o = (wave * 3 + t) * 16 + lo;
    float db = cb[o];
    float sc = g[o] * rsqrtf(vr[o] + LN_EPS);
    float sh = bb[o] - mn[o] * sc;
#pragma unroll
    for (int r = 0; r < 8; ++r)
      sT[hi * 8 + r][o] = fmaxf((acc[t][r] + db) * sc + sh, 0.0f);
  }
  __syncthreads();
  // final 1x1 conv: [16,384] x out_w^T -> [16,15]
  if (tid < 240) {
    int r = tid / 15, o15 = tid % 15;
    float s = ob[o15];
    const float* wrow = ow + (size_t)o15 * 384;
    for (int c = 0; c < 384; ++c) s += sT[r][c] * wrow[c];
    int grow = rowBase + r;
    int b = grow >> 12, n = grow & 4095;
    int hh = n >> 8, w2 = (n >> 4) & 15, dd = n & 15;
    size_t oi = ((size_t)b * 15 + o15) * 32768 +
                (size_t)(2 * hh + i2) * 1024 + (2 * w2 + j2) * 32 + (2 * dd + k2);
    out[oi] = s;
  }
}

// ---------------------------------------------------------------------------
// Host launcher
// ---------------------------------------------------------------------------
extern "C" void kernel_launch(void* const* d_in, const int* in_sizes, int n_in,
                              void* d_out, int out_size, void* d_ws, size_t ws_size,
                              hipStream_t stream) {
  (void)in_sizes; (void)n_in; (void)out_size; (void)ws_size;
  const float* x       = (const float*)d_in[0];
  const float* patch_w = (const float*)d_in[1];
  const float* patch_b = (const float*)d_in[2];
  const float* pos_emb = (const float*)d_in[3];
  const float* m_ln1_g = (const float*)d_in[4];
  const float* m_ln1_b = (const float*)d_in[5];
  const float* m_qw    = (const float*)d_in[6];
  const float* m_kw    = (const float*)d_in[7];
  const float* m_vw    = (const float*)d_in[8];
  const float* m_pjw   = (const float*)d_in[9];
  const float* m_pjb   = (const float*)d_in[10];
  const float* m_ln2_g = (const float*)d_in[11];
  const float* m_ln2_b = (const float*)d_in[12];
  const float* m_w1    = (const float*)d_in[13];
  const float* m_b1    = (const float*)d_in[14];
  const float* m_w2    = (const float*)d_in[15];
  const float* m_b2    = (const float*)d_in[16];
  const float* s_ln1_g = (const float*)d_in[17];
  const float* s_ln1_b = (const float*)d_in[18];
  const float* s_qkv_w = (const float*)d_in[19];
  const float* s_qkv_b = (const float*)d_in[20];
  const float* s_pjw   = (const float*)d_in[21];
  const float* s_pjb   = (const float*)d_in[22];
  const float* s_ln2_g = (const float*)d_in[23];
  const float* s_ln2_b = (const float*)d_in[24];
  const float* s_w1    = (const float*)d_in[25];
  const float* s_b1    = (const float*)d_in[26];
  const float* s_w2    = (const float*)d_in[27];
  const float* s_b2    = (const float*)d_in[28];
  const float* d1_w    = (const float*)d_in[29];
  const float* d1_b    = (const float*)d_in[30];
  const float* bn1_g   = (const float*)d_in[31];
  const float* bn1_b   = (const float*)d_in[32];
  const float* bn1_m   = (const float*)d_in[33];
  const float* bn1_v   = (const float*)d_in[34];
  const float* d2_w    = (const float*)d_in[35];
  const float* d2_b    = (const float*)d_in[36];
  const float* bn2_g   = (const float*)d_in[37];
  const float* bn2_b   = (const float*)d_in[38];
  const float* bn2_m   = (const float*)d_in[39];
  const float* bn2_v   = (const float*)d_in[40];
  const float* out_w   = (const float*)d_in[41];
  const float* out_b   = (const float*)d_in[42];

  // --- bf16 weight arena layout (elements) ---
  unsigned short* BF = (unsigned short*)d_ws;
  size_t off = 0;
  size_t oPatch = off; off += 512 * 384;
  size_t oQKVd[4], oProjd[4], oW1d[4], oW2d[4];
  size_t oQKVs[4], oProjs[4], oW1s[4], oW2s[4];
  for (int j = 0; j < 4; ++j) {
    oQKVd[j]  = off; off += 384 * 1152;
    oProjd[j] = off; off += 384 * 384;
    oW1d[j]   = off; off += 384 * 1536;
    oW2d[j]   = off; off += 1536 * 384;
    oQKVs[j]  = off; off += 384 * 1152;
    oProjs[j] = off; off += 384 * 384;
    oW1s[j]   = off; off += 384 * 1536;
    oW2s[j]   = off; off += 1536 * 384;
  }
  size_t oD1 = off; off += 384 * 3072;
  size_t oD2 = off; off += 384 * 3072;
  size_t bfBytes = (off * 2 + 255) & ~(size_t)255;

  // --- f32 activation buffers ---
  float* F    = (float*)((char*)d_ws + bfBytes);
  float* fIm  = F;                 // [2048, 512]
  float* fT   = fIm + 2048 * 512;  // [2048, 384]
  float* fXn  = fT + 2048 * 384;   // [2048, 384]
  float* fQkv = fXn + 2048 * 384;  // [2048, 1152]
  float* fS   = fQkv + 2048 * 1152;        // [24, 512, 512]
  float* fO   = fS + 24 * 512 * 512;       // [2048, 384]
  float* fH   = fO + 2048 * 384;           // [2048, 1536]
  float* fR1  = fH + 2048 * 1536;          // [16384, 384]

  // --- weight conversion ---
  cvt_patchT_k<<<768, 256, 0, stream>>>(patch_w, BF + oPatch);
  for (int j = 0; j < 4; ++j) {
    cvt_qkv3_k<<<1728, 256, 0, stream>>>(m_qw + (size_t)j * 147456,
                                         m_kw + (size_t)j * 147456,
                                         m_vw + (size_t)j * 147456, BF + oQKVd[j]);
    cvt_mat_k<<<576, 256, 0, stream>>>(m_pjw + (size_t)j * 147456, BF + oProjd[j], 384, 384);
    cvt_mat_k<<<2304, 256, 0, stream>>>(m_w1 + (size_t)j * 589824, BF + oW1d[j], 384, 1536);
    cvt_mat_k<<<2304, 256, 0, stream>>>(m_w2 + (size_t)j * 589824, BF + oW2d[j], 1536, 384);
    cvt_mat_k<<<1728, 256, 0, stream>>>(s_qkv_w + (size_t)j * 442368, BF + oQKVs[j], 384, 1152);
    cvt_mat_k<<<576, 256, 0, stream>>>(s_pjw + (size_t)j * 147456, BF + oProjs[j], 384, 384);
    cvt_mat_k<<<2304, 256, 0, stream>>>(s_w1 + (size_t)j * 589824, BF + oW1s[j], 384, 1536);
    cvt_mat_k<<<2304, 256, 0, stream>>>(s_w2 + (size_t)j * 589824, BF + oW2s[j], 1536, 384);
  }
  cvt_deconv_k<<<4608, 256, 0, stream>>>(d1_w, BF + oD1);
  cvt_deconv_k<<<4608, 256, 0, stream>>>(d2_w, BF + oD2);

  // --- patch embedding + pos_emb (res broadcast) ---
  im2col_k<<<2048, 256, 0, stream>>>(x, fIm);
  gemm_t<2, 0, 1><<<dim3(3, 32), 256, 0, stream>>>(fIm, 512, BF + oPatch, patch_b,
                                                   pos_emb, 384, fT, 384, 512);

  // --- 8 transformer layers ---
  for (int L = 0; L < 8; ++L) {
    int j = L >> 1;
    if ((L & 1) == 0) {  // dense block
      layernorm_k<<<256, 256, 0, stream>>>(fT, m_ln1_g + j * 384, m_ln1_b + j * 384, fXn);
      gemm_t<0, 0, 0><<<dim3(9, 32), 256, 0, stream>>>(fXn, 384, BF + oQKVd[j], nullptr,
                                                       nullptr, 0, fQkv, 1152, 384);
      attn_qk_k<<<dim3(8, 8, 24), 256, 0, stream>>>(fQkv, fS);
      softmax512_k<<<1536, 256, 0, stream>>>(fS);
      attn_pv_k<<<dim3(8, 24), 256, 0, stream>>>(fS, fQkv, fO);
      gemm_t<1, 0, 1><<<dim3(3, 32), 256, 0, stream>>>(fO, 384, BF + oProjd[j], m_pjb + j * 384,
                                                       fT, 384, fT, 384, 384);
      layernorm_k<<<256, 256, 0, stream>>>(fT, m_ln2_g + j * 384, m_ln2_b + j * 384, fXn);
      gemm_t<0, 1, 1><<<dim3(12, 32), 256, 0, stream>>>(fXn, 384, BF + oW1d[j], m_b1 + j * 1536,
                                                        nullptr, 0, fH, 1536, 384);
      gemm_t<1, 0, 1><<<dim3(3, 32), 256, 0, stream>>>(fH, 1536, BF + oW2d[j], m_b2 + j * 384,
                                                       fT, 384, fT, 384, 1536);
    } else {             // shifted-window block
      layernorm_k<<<256, 256, 0, stream>>>(fT, s_ln1_g + j * 384, s_ln1_b + j * 384, fXn);
      gemm_t<0, 0, 1><<<dim3(9, 32), 256, 0, stream>>>(fXn, 384, BF + oQKVs[j], s_qkv_b + j * 1152,
                                                       nullptr, 0, fQkv, 1152, 384);
      swin_attn_k<<<192, 256, 0, stream>>>(fQkv, fO);
      gemm_t<1, 0, 1><<<dim3(3, 32), 256, 0, stream>>>(fO, 384, BF + oProjs[j], s_pjb + j * 384,
                                                       fT, 384, fT, 384, 384);
      layernorm_k<<<256, 256, 0, stream>>>(fT, s_ln2_g + j * 384, s_ln2_b + j * 384, fXn);
      gemm_t<0, 1, 1><<<dim3(12, 32), 256, 0, stream>>>(fXn, 384, BF + oW1s[j], s_b1 + j * 1536,
                                                        nullptr, 0, fH, 1536, 384);
      gemm_t<1, 0, 1><<<dim3(3, 32), 256, 0, stream>>>(fH, 1536, BF + oW2s[j], s_b2 + j * 384,
                                                       fT, 384, fT, 384, 1536);
    }
  }

  // --- decoder ---
  deconv1_k<<<dim3(128, 8), 256, 0, stream>>>(fT, BF + oD1, d1_b, bn1_g, bn1_b, bn1_m, bn1_v, fR1);
  deconv2_out_k<<<dim3(1024, 8), 256, 0, stream>>>(fR1, BF + oD2, d2_b, bn2_g, bn2_b,
                                                   bn2_m, bn2_v, out_w, out_b, (float*)d_out);
}
